// GNNRes_19164144075221
// MI455X (gfx1250) — matrix operand order repaired and checked by reading, not compiled, not run on hardware
//
#include <hip/hip_runtime.h>
#include <hip/hip_bf16.h>

#define BATCH 64
#define NN    1024
#define CC    64

typedef __attribute__((ext_vector_type(16))) __bf16 v16bf;
typedef __attribute__((ext_vector_type(8)))  float  v8f;

__device__ __forceinline__ unsigned short f2bf(float f) {
  unsigned int u = __float_as_uint(f);
  u += 0x7FFFu + ((u >> 16) & 1u);          // round-to-nearest-even
  return (unsigned short)(u >> 16);
}
__device__ __forceinline__ float bf2f(unsigned short h) {
  return __uint_as_float((unsigned int)h << 16);
}
__device__ __forceinline__ float wave_sum(float v) {
  #pragma unroll
  for (int o = 16; o > 0; o >>= 1) v += __shfl_xor(v, o, 32);
  return v;
}

// ---------------- per-batch normalization of e ----------------
__global__ void k_norm(const float* __restrict__ e, float* __restrict__ x0) {
  __shared__ float red[2][8];
  int b = blockIdx.x, t = threadIdx.x;
  const float* eb = e + (size_t)b * NN;
  float s = 0.f, s2 = 0.f;
  for (int n = t; n < NN; n += 256) { float v = eb[n]; s += v; s2 += v * v; }
  s = wave_sum(s); s2 = wave_sum(s2);
  int lane = t & 31, w = t >> 5;
  if (lane == 0) { red[0][w] = s; red[1][w] = s2; }
  __syncthreads();
  if (t == 0) {
    float ts = 0.f, ts2 = 0.f;
    for (int i = 0; i < 8; ++i) { ts += red[0][i]; ts2 += red[1][i]; }
    red[0][0] = ts; red[1][0] = ts2;
  }
  __syncthreads();
  float mean = red[0][0] / NN;
  float var  = red[1][0] / NN - mean * mean;
  float inv  = 1.f / (sqrtf(fmaxf(var, 0.f)) + 1e-5f);
  for (int n = t; n < NN; n += 256) x0[(size_t)b * NN + n] = (eb[n] - mean) * inv;
}

// ---------------- unnormalized Gaussian kernel matrix E (bf16) + 1/rowsum ----
__global__ void k_adj(const float* __restrict__ phi, const float* __restrict__ eta,
                      unsigned short* __restrict__ E, float* __restrict__ inv_s) {
  __shared__ float ph[NN], et[NN];
  int b = blockIdx.x, t = threadIdx.x;
  for (int n = t; n < NN; n += 256) {
    ph[n] = phi[(size_t)b * NN + n];
    et[n] = eta[(size_t)b * NN + n];
  }
  __syncthreads();
  int lane = t & 31, w = t >> 5;
  int n = blockIdx.y * 8 + w;                   // one wave per adjacency row
  float pn = ph[n], en = et[n];
  unsigned short* Er = E + ((size_t)b * NN + n) * NN;
  float sum = 0.f;
  for (int m = lane; m < NN; m += 32) {
    float dp = pn - ph[m], de = en - et[m];
    float v  = expf(-(dp * dp + de * de));
    sum += v;
    Er[m] = f2bf(v);
  }
  sum = wave_sum(sum);
  if (lane == 0) inv_s[(size_t)b * NN + n] = 1.f / sum;   // >= 1, safe
}

// ---------------- layer-0 propagation (cin = 1): p0 = (x0/s) @ E ------------
__global__ void k_matvec0(const float* __restrict__ x0, const float* __restrict__ inv_s,
                          const unsigned short* __restrict__ E, float* __restrict__ p0) {
  __shared__ float xs[NN];
  int b = blockIdx.x, t = threadIdx.x;
  for (int n = t; n < NN; n += 256)
    xs[n] = x0[(size_t)b * NN + n] * inv_s[(size_t)b * NN + n];
  __syncthreads();
  int lane = t & 31, w = t >> 5;
  int m = blockIdx.y * 8 + w;
  const unsigned short* Er = E + ((size_t)b * NN + m) * NN;   // E symmetric
  float acc = 0.f;
  for (int n = lane; n < NN; n += 32) acc += xs[n] * bf2f(Er[n]);
  acc = wave_sum(acc);
  if (lane == 0) p0[(size_t)b * NN + m] = acc;
}

// ---------------- layer-0 channel mix: X1 + pre-scaled bf16 Xs --------------
__global__ void k_mix0(const float* __restrict__ x0, const float* __restrict__ p0,
                       const float* __restrict__ W0, const float* __restrict__ b0,
                       const float* __restrict__ Wsc0, const float* __restrict__ inv_s,
                       float* __restrict__ X, unsigned short* __restrict__ Xs) {
  int tid = blockIdx.x * 256 + threadIdx.x;     // tid = b*NN + m
  int b = tid >> 10, m = tid & (NN - 1);
  float x = x0[tid], p = p0[tid], iv = inv_s[tid];
  size_t base = (size_t)b * CC * NN + m;
  #pragma unroll 4
  for (int c = 0; c < CC; ++c) {
    float v = fmaxf(fmaf(W0[2 * c], x, fmaf(W0[2 * c + 1], p, b0[c])), 0.f) + Wsc0[c] * x;
    X[base + (size_t)c * NN]  = v;
    Xs[base + (size_t)c * NN] = f2bf(v * iv);
  }
}

// ---------------- WMMA propagation: P = Xs @ E  (bf16 x bf16 -> f32) --------
// Block = 128 threads (4 waves). Output tile 64(c) x 64(m). K loop over 1024.
#define LSTRIDE 40   // bf16 elements per LDS row: 80B => 16B aligned, conflict-free
__global__ __launch_bounds__(128) void k_prop(const unsigned short* __restrict__ Xs,
                                              const unsigned short* __restrict__ E,
                                              float* __restrict__ P) {
  __shared__ unsigned short As[64 * LSTRIDE];   // Xs tile: [c][k]
  __shared__ unsigned short Bs[64 * LSTRIDE];   // E  tile: [m][k]  (E symmetric)
  int b  = blockIdx.x;
  int m0 = blockIdx.y * 64;
  int t  = threadIdx.x;
  int lane = t & 31, wave = t >> 5;
  int l15 = lane & 15, hi = lane >> 4;
  int cbase = wave * 16;

  const unsigned short* Xb = Xs + (size_t)b * CC * NN;
  const unsigned short* Eb = E  + (size_t)b * NN * NN;

  v8f acc[4] = {};

  for (int kt = 0; kt < NN; kt += 32) {
    __syncthreads();
    // stage two 64x32 bf16 tiles; 256 16B-chunks each, 2 per thread
    #pragma unroll
    for (int i = 0; i < 2; ++i) {
      int idx = t + 128 * i;
      int row = idx >> 2, seg = idx & 3;
      *(uint4*)&As[row * LSTRIDE + seg * 8] =
          *(const uint4*)(Xb + (size_t)row * NN + kt + seg * 8);
      *(uint4*)&Bs[row * LSTRIDE + seg * 8] =
          *(const uint4*)(Eb + (size_t)(m0 + row) * NN + kt + seg * 8);
      if (kt + 32 < NN) {
        __builtin_prefetch((const void*)(Xb + (size_t)row * NN + kt + 32 + seg * 8), 0, 1);
        __builtin_prefetch((const void*)(Eb + (size_t)(m0 + row) * NN + kt + 32 + seg * 8), 0, 1);
      }
    }
    __syncthreads();

    // A fragment: 16x32 bf16 per ISA layout (lane row = l15, K pairs per VGPR)
    union { unsigned int u[8]; v16bf v; } af;
    #pragma unroll
    for (int r = 0; r < 8; ++r) {
      int k0 = ((r >> 2) << 4) + (hi << 3) + ((r & 3) << 1);
      af.u[r] = *(const unsigned int*)&As[(cbase + l15) * LSTRIDE + k0];
    }
    // 4 B fragments (m sub-tiles), B is 32x16: same K order as A (K=16*(r>>2)+8*hi+2*(r&3))
    #pragma unroll
    for (int j = 0; j < 4; ++j) {
      union { unsigned int u[8]; v16bf v; } bfrg;
      #pragma unroll
      for (int r = 0; r < 8; ++r) {
        int k0 = ((r >> 2) << 4) + (hi << 3) + ((r & 3) << 1);
        bfrg.u[r] = *(const unsigned int*)&Bs[((j << 4) + l15) * LSTRIDE + k0];
      }
      acc[j] = __builtin_amdgcn_wmma_f32_16x16x32_bf16(
          false, af.v, false, bfrg.v, (short)0, acc[j], false, false);
    }
  }

  // epilogue: C/D layout — VGPR r holds row (cbase + r + 8*hi), lane col = l15
  float* Pb = P + (size_t)b * CC * NN;
  #pragma unroll
  for (int j = 0; j < 4; ++j) {
    int m = m0 + (j << 4) + l15;
    #pragma unroll
    for (int r = 0; r < 8; ++r) {
      int c = cbase + r + (hi << 3);
      Pb[(size_t)c * NN + m] = acc[j][r];
    }
  }
}

// ---------------- generic channel mix: Xout = relu(W@[X;P]+b) + Wsc@X -------
__global__ void k_mix(const float* __restrict__ X, const float* __restrict__ P,
                      const float* __restrict__ W, const float* __restrict__ bias,
                      const float* __restrict__ Wsc, const float* __restrict__ inv_s,
                      float* __restrict__ Xout, unsigned short* __restrict__ Xs,
                      int writeXs) {
  __shared__ float Xt[64 * 65];
  __shared__ float Pt[64 * 65];
  int b = blockIdx.x, m0 = blockIdx.y * 64, t = threadIdx.x;
  size_t gbase = (size_t)b * CC * NN + m0;
  #pragma unroll
  for (int i = 0; i < 16; ++i) {
    int idx = t + 256 * i;
    int c = idx >> 6, j = idx & 63;
    Xt[c * 65 + j] = X[gbase + (size_t)c * NN + j];
    Pt[c * 65 + j] = P[gbase + (size_t)c * NN + j];
  }
  __syncthreads();
  int j  = t & 63;
  int cg = t >> 6;                 // uniform within a wave32
  float iv = inv_s[(size_t)b * NN + m0 + j];
  for (int cc = 0; cc < 16; ++cc) {
    int c = cg * 16 + cc;
    const float* Wr  = W   + c * 128;   // scalar (uniform) loads
    const float* Wsr = Wsc + c * 64;
    float acc = bias[c], sc = 0.f;
    #pragma unroll 8
    for (int k = 0; k < 64; ++k) {
      float xv = Xt[k * 65 + j];
      acc = fmaf(Wr[k],      xv,             acc);
      acc = fmaf(Wr[64 + k], Pt[k * 65 + j], acc);
      sc  = fmaf(Wsr[k],     xv,             sc);
    }
    float v = fmaxf(acc, 0.f) + sc;
    Xout[gbase + (size_t)c * NN + j] = v;
    if (writeXs) Xs[gbase + (size_t)c * NN + j] = f2bf(v * iv);
  }
}

// ---------------- final: row sums (propagation eliminated) + sigmoid --------
__global__ void k_final(const float* __restrict__ X3, const float* __restrict__ Wf,
                        const float* __restrict__ bfin, float* __restrict__ out) {
  __shared__ float srow[CC];
  __shared__ float part[CC];
  int b = blockIdx.x, t = threadIdx.x;
  int lane = t & 31, w = t >> 5;
  const float* Xb = X3 + (size_t)b * CC * NN;
  for (int kk = 0; kk < CC / 8; ++kk) {
    int k = w * 8 + kk;
    float s = 0.f;
    for (int n = lane; n < NN; n += 32) s += Xb[(size_t)k * NN + n];
    s = wave_sum(s);
    if (lane == 0) srow[k] = s;
  }
  __syncthreads();
  if (t < CC) part[t] = (Wf[t] + Wf[CC + t]) * srow[t];  // adj row-stochastic
  __syncthreads();
  if (t == 0) {
    float tot = 0.f;
    for (int k = 0; k < CC; ++k) tot += part[k];
    float z = tot / NN + bfin[0];
    out[b] = 1.f / (1.f + expf(-z));
  }
}

extern "C" void kernel_launch(void* const* d_in, const int* in_sizes, int n_in,
                              void* d_out, int out_size, void* d_ws, size_t ws_size,
                              hipStream_t stream) {
  (void)in_sizes; (void)n_in; (void)out_size; (void)ws_size;
  const float* e    = (const float*)d_in[0];
  const float* phi  = (const float*)d_in[1];
  const float* eta  = (const float*)d_in[2];
  const float* W0   = (const float*)d_in[3];
  const float* b0   = (const float*)d_in[4];
  const float* Wsc0 = (const float*)d_in[5];
  const float* W1   = (const float*)d_in[6];
  const float* b1   = (const float*)d_in[7];
  const float* Wsc1 = (const float*)d_in[8];
  const float* W2   = (const float*)d_in[9];
  const float* b2   = (const float*)d_in[10];
  const float* Wsc2 = (const float*)d_in[11];
  const float* Wf   = (const float*)d_in[12];
  const float* bfin = (const float*)d_in[13];
  float* out = (float*)d_out;

  char* ws = (char*)d_ws;
  size_t off = 0;
  unsigned short* E  = (unsigned short*)(ws + off); off += (size_t)BATCH * NN * NN * 2; // 128 MiB
  float* inv_s = (float*)(ws + off); off += (size_t)BATCH * NN * 4;
  float* x0    = (float*)(ws + off); off += (size_t)BATCH * NN * 4;
  float* p0    = (float*)(ws + off); off += (size_t)BATCH * NN * 4;
  float* XA    = (float*)(ws + off); off += (size_t)BATCH * CC * NN * 4;
  float* XB    = (float*)(ws + off); off += (size_t)BATCH * CC * NN * 4;
  float* P     = (float*)(ws + off); off += (size_t)BATCH * CC * NN * 4;
  unsigned short* Xs = (unsigned short*)(ws + off); off += (size_t)BATCH * CC * NN * 2;

  k_norm   <<<BATCH, 256, 0, stream>>>(e, x0);
  k_adj    <<<dim3(BATCH, NN / 8), 256, 0, stream>>>(phi, eta, E, inv_s);
  k_matvec0<<<dim3(BATCH, NN / 8), 256, 0, stream>>>(x0, inv_s, E, p0);
  k_mix0   <<<(BATCH * NN) / 256, 256, 0, stream>>>(x0, p0, W0, b0, Wsc0, inv_s, XA, Xs);
  k_prop   <<<dim3(BATCH, NN / 64), 128, 0, stream>>>(Xs, E, P);
  k_mix    <<<dim3(BATCH, NN / 64), 256, 0, stream>>>(XA, P, W1, b1, Wsc1, inv_s, XB, Xs, 1);
  k_prop   <<<dim3(BATCH, NN / 64), 128, 0, stream>>>(Xs, E, P);
  k_mix    <<<dim3(BATCH, NN / 64), 256, 0, stream>>>(XB, P, W2, b2, Wsc2, inv_s, XA, Xs, 0);
  k_final  <<<BATCH, 256, 0, stream>>>(XA, Wf, bfin, out);
}
